// SequenceLSTMDecoder_78563541778994
// MI455X (gfx1250) — matrix operand order repaired and checked by reading, not compiled, not run on hardware
//
#include <hip/hip_runtime.h>

// ---------------------------------------------------------------------------
// SequenceLSTMDecoder for MI455X (gfx1250, wave32, WMMA bf16 16x16x32 / f32 acc)
// B=16 (== WMMA M tile), S=256, H=512, V=32000
// ---------------------------------------------------------------------------

typedef __attribute__((ext_vector_type(16))) __bf16        v16bf;
typedef __attribute__((ext_vector_type(8)))  float         v8f;
typedef __attribute__((ext_vector_type(4)))  float         v4f;
typedef __attribute__((ext_vector_type(4)))  unsigned int  v4u;

#define BDIM 16
#define SDIM 256
#define HDIM 512
#define VDIM 32000
#define G4H  2048           // 4*H
#define NTG  128            // 2048/16 gate-column tiles
#define KC   16             // 512/32 K chunks
#define VT   2000           // 32000/16 vocab tiles
#define START_TOK 1

// k_logits blocking
#define MTW 2               // mt tiles per wave (register blocking)
#define NTB 4               // nt tiles per block (B panel shared via LDS)
#define WPB 8               // waves per block
#define MT_PER_BLK (WPB * MTW)          // 16
#define LOGITS_LDS (NTB * 16 * HDIM * 2)  // 64 KB

static __device__ __forceinline__ unsigned short f2bf(float f) {
    unsigned u = __builtin_bit_cast(unsigned, f);
    unsigned r = u + 0x7FFFu + ((u >> 16) & 1u);   // round-to-nearest-even
    return (unsigned short)(r >> 16);
}

static __device__ __forceinline__ float sigm(float x) {
    return 1.0f / (1.0f + __expf(-x));
}

// A-matrix fragment (16x32 bf16, M x K).  Lane L<16 holds row L,
// K in {0..15}; lane L>=16 holds row L-16, K in {16..31} (matches B order).
static __device__ __forceinline__ v16bf load_a_frag(const unsigned short* __restrict__ base, int k0) {
    int lane = threadIdx.x & 31;
    int row  = lane & 15;
    int hi   = lane >> 4;
    const unsigned short* p = base + row * HDIM + k0 + hi * 16;
    union { v16bf v; v4u u[2]; } r;
    r.u[0] = *(const v4u*)p;            // K = k0+hi*16   .. +7  (16 B)
    r.u[1] = *(const v4u*)(p + 8);      // K = k0+hi*16+8 .. +7  (16 B)
    return r.v;
}

// B-matrix fragment (32x16 bf16, K x N) from a row-major [N][K] weight matrix
// (B[k][n] = W[ncol0+n][k]).  Lane L holds column n=L&15, a contiguous 16-elem
// K run starting at k0 + (L>=16 ? 16 : 0).
static __device__ __forceinline__ v16bf load_b_frag(const unsigned short* __restrict__ wrowmaj,
                                                    int ncol0, int k0) {
    int lane = threadIdx.x & 31;
    int n    = lane & 15;
    int hi   = lane >> 4;
    const unsigned short* p = wrowmaj + (size_t)(ncol0 + n) * HDIM + k0 + hi * 16;
    union { v16bf v; v4u u[2]; } r;
    r.u[0] = *(const v4u*)p;
    r.u[1] = *(const v4u*)(p + 8);
    return r.v;
}

// Same B fragment but sourced from an LDS panel of [NTB*16][HDIM] bf16.
static __device__ __forceinline__ v16bf load_b_frag_lds(const unsigned short* ldsb,
                                                        int ntile_local, int k0) {
    int lane = threadIdx.x & 31;
    int n    = lane & 15;
    int hi   = lane >> 4;
    const unsigned short* p = ldsb + (ntile_local * 16 + n) * HDIM + k0 + hi * 16;
    union { v16bf v; v4u u[2]; } r;
    r.u[0] = *(const v4u*)p;            // ds_load_b128
    r.u[1] = *(const v4u*)(p + 8);
    return r.v;
}

static __device__ __forceinline__ v8f wmma_bf16(v16bf a, v16bf b, v8f c) {
    return __builtin_amdgcn_wmma_f32_16x16x32_bf16(false, a, false, b, (short)0, c, false, false);
}

// ---------------------------------------------------------------------------
// Prep kernels
// ---------------------------------------------------------------------------

__global__ void k_cvt_bf16(const float* __restrict__ src, unsigned short* __restrict__ dst, int npairs) {
    int i = blockIdx.x * blockDim.x + threadIdx.x;
    int stride = gridDim.x * blockDim.x;
    for (; i < npairs; i += stride) {
        unsigned a = f2bf(src[2 * i]);
        unsigned b = f2bf(src[2 * i + 1]);
        ((unsigned*)dst)[i] = a | (b << 16);
    }
}

__global__ void k_bias(const float* __restrict__ bih, const float* __restrict__ bhh,
                       float* __restrict__ bsum) {
    int i = blockIdx.x * blockDim.x + threadIdx.x;
    if (i < G4H) bsum[i] = bih[i] + bhh[i];
}

__global__ void k_h0(const float* __restrict__ h0, unsigned short* __restrict__ hcur) {
    int i = blockIdx.x * blockDim.x + threadIdx.x;
    if (i < BDIM * HDIM) hcur[i] = f2bf(h0[i]);
}

__global__ void k_init_ctr(unsigned int* ctr) {
    if (threadIdx.x == 0) *ctr = 0u;
}

// Gather teacher-forced token embeddings into row order (s*16 + b).
__global__ void k_gather(const int* __restrict__ tgt, const unsigned short* __restrict__ emb16,
                         unsigned short* __restrict__ eseq) {
    int row = blockIdx.x;                 // 0..4095 == s*16 + b
    int s = row >> 4, b = row & 15;
    int tok = (s == 0) ? START_TOK : tgt[b * SDIM + (s - 1)];
    const unsigned* src = (const unsigned*)(emb16 + (size_t)tok * HDIM);
    unsigned*       dst = (unsigned*)(eseq + (size_t)row * HDIM);
    dst[threadIdx.x] = src[threadIdx.x];  // 256 dwords == 512 bf16
}

// ---------------------------------------------------------------------------
// xW = emb @ W_ih^T + (b_ih + b_hh), stored fragment-native:
//   frag index fi = s*128 + ntile ; float8 at xw[(fi*32 + lane)*8]
// ---------------------------------------------------------------------------
__global__ void k_xw(const unsigned short* __restrict__ eseq, const unsigned short* __restrict__ wih16,
                     const float* __restrict__ bsum, float* __restrict__ xw) {
    int w    = blockIdx.x * (blockDim.x >> 5) + (threadIdx.x >> 5);  // tile id
    int lane = threadIdx.x & 31;
    int mt = w >> 7;            // == s
    int nt = w & (NTG - 1);     // gate-col tile
    int n  = lane & 15;

    float bv = bsum[nt * 16 + n];
    v8f acc;
#pragma unroll
    for (int r = 0; r < 8; r++) acc[r] = bv;

    const unsigned short* A = eseq + (size_t)mt * 16 * HDIM;
#pragma unroll 4
    for (int kc = 0; kc < KC; kc++) {
        v16bf a = load_a_frag(A, kc * 32);
        v16bf b = load_b_frag(wih16, nt * 16, kc * 32);
        acc = wmma_bf16(a, b, acc);
    }

    float* p = xw + ((size_t)w * 32 + lane) * 8;
    union { v8f v; v4f q[2]; } u; u.v = acc;
    *(v4f*)p       = u.q[0];
    *(v4f*)(p + 4) = u.q[1];
}

// ---------------------------------------------------------------------------
// Persistent LSTM recurrence: 16 blocks x 2 waves = 32 waves, one 16-wide
// h-column tile each.  Per step: 4 gate tiles x 16 K-chunk WMMAs; grid barrier
// via monotonic atomic counter; h exchanged via double-buffered global bf16.
// ---------------------------------------------------------------------------
__global__ void k_lstm(const float* __restrict__ c0, const float* __restrict__ xw,
                       const unsigned short* __restrict__ whh16,
                       unsigned short* hA, unsigned short* hB,
                       unsigned short* __restrict__ hs16, unsigned int* ctr) {
    const int lane = threadIdx.x & 31;
    const int t    = blockIdx.x * (blockDim.x >> 5) + (threadIdx.x >> 5);  // 0..31
    const int n    = lane & 15;
    const int hi   = lane >> 4;
    const int col  = t * 16 + n;

    // cell state tile in WMMA D layout (lane l, reg r -> row r+8*hi, col)
    v8f c;
#pragma unroll
    for (int r = 0; r < 8; r++) c[r] = c0[(r + 8 * hi) * HDIM + col];

    for (int s = 0; s < SDIM; s++) {
        const unsigned short* hin  = (s & 1) ? hB : hA;
        unsigned short*       hout = (s & 1) ? hA : hB;

        // init gate accumulators from precomputed xW fragments
        v8f acc[4];
#pragma unroll
        for (int g = 0; g < 4; g++) {
            const float* p = xw + ((size_t)(s * NTG + g * 32 + t) * 32 + lane) * 8;
            union { v8f v; v4f q[2]; } u;
            u.q[0] = *(const v4f*)p;
            u.q[1] = *(const v4f*)(p + 4);
            acc[g] = u.v;
        }
        if (s + 1 < SDIM)   // global_prefetch_b8 of next step's xW fragment
            __builtin_prefetch(xw + ((size_t)((s + 1) * NTG + t) * 32 + lane) * 8, 0, 1);

#pragma unroll 2
        for (int kc = 0; kc < KC; kc++) {
            v16bf a = load_a_frag(hin, kc * 32);
#pragma unroll
            for (int g = 0; g < 4; g++) {
                v16bf b = load_b_frag(whh16, g * HDIM + t * 16, kc * 32);
                acc[g] = wmma_bf16(a, b, acc[g]);
            }
        }

        // pointwise (all 4 gate frags share the (row,col) lane mapping)
        v8f hv;
#pragma unroll
        for (int r = 0; r < 8; r++) {
            float iv = sigm(acc[0][r]);
            float fv = sigm(acc[1][r]);
            float gv = tanhf(acc[2][r]);
            float ov = sigm(acc[3][r]);
            float cv = fv * c[r] + iv * gv;
            c[r]  = cv;
            hv[r] = ov * tanhf(cv);
        }
#pragma unroll
        for (int r = 0; r < 8; r++) {
            unsigned short hb = f2bf(hv[r]);
            int row = r + 8 * hi;
            hout[row * HDIM + col] = hb;                         // next-step input
            hs16[((size_t)s * 16 + row) * HDIM + col] = hb;      // projection A row s*16+b
        }

        // ---- grid barrier (16 blocks, monotonic counter) ----
        __threadfence();
        __syncthreads();
        if (threadIdx.x == 0) {
            atomicAdd(ctr, 1u);
            unsigned want = (unsigned)gridDim.x * (unsigned)(s + 1);
            volatile unsigned* vc = ctr;
            while (*vc < want) __builtin_amdgcn_s_sleep(1);
        }
        __syncthreads();
        __threadfence();
    }
}

// ---------------------------------------------------------------------------
// logits = hs @ embedding^T + out_bias (tied weights), 524 MB f32 output.
// Block: 8 waves share a 4-tile (64-row) B panel staged in 64 KB LDS; each
// wave register-blocks a 2(mt) x 4(nt) super-tile (A frags reused 4x, B 2x).
// ---------------------------------------------------------------------------
__global__ void __launch_bounds__(256)
k_logits(const unsigned short* __restrict__ hs16,
         const unsigned short* __restrict__ emb16,
         const float* __restrict__ obias, float* __restrict__ out) {
    extern __shared__ unsigned short ldsb[];   // [NTB*16][HDIM] bf16 = 64 KB

    const int wave = threadIdx.x >> 5;
    const int lane = threadIdx.x & 31;
    const int n    = lane & 15;
    const int hi   = lane >> 4;

    const int mtg = blockIdx.x & (SDIM / MT_PER_BLK - 1);   // 16 mt-groups
    const int ntg = blockIdx.x / (SDIM / MT_PER_BLK);       // 500 nt-groups
    const int mt0 = mtg * MT_PER_BLK + wave * MTW;
    const int nt0 = ntg * NTB;

    // Stage B panel: 64 consecutive embedding rows (nt0*16 ..), fully
    // contiguous in emb16 -> coalesced b128 copy into LDS.
    {
        const v4u* src = (const v4u*)(emb16 + (size_t)nt0 * 16 * HDIM);
        v4u* dst = (v4u*)ldsb;
        const int total = (NTB * 16 * HDIM) / 8;   // uint4 count = 4096
        for (int i = threadIdx.x; i < total; i += blockDim.x) dst[i] = src[i];
    }
    __syncthreads();

    v8f acc[MTW][NTB];
#pragma unroll
    for (int mi = 0; mi < MTW; mi++)
#pragma unroll
        for (int ni = 0; ni < NTB; ni++) {
            float bv = obias[(nt0 + ni) * 16 + n];
#pragma unroll
            for (int r = 0; r < 8; r++) acc[mi][ni][r] = bv;
        }

    const unsigned short* A0 = hs16 + (size_t)mt0 * 16 * HDIM;
    for (int kc = 0; kc < KC; kc++) {
        v16bf bfrag[NTB];
#pragma unroll
        for (int ni = 0; ni < NTB; ni++)
            bfrag[ni] = load_b_frag_lds(ldsb, ni, kc * 32);
#pragma unroll
        for (int mi = 0; mi < MTW; mi++) {
            v16bf a = load_a_frag(A0 + (size_t)mi * 16 * HDIM, kc * 32);
#pragma unroll
            for (int ni = 0; ni < NTB; ni++)
                acc[mi][ni] = wmma_bf16(a, bfrag[ni], acc[mi][ni]);
        }
    }

#pragma unroll
    for (int mi = 0; mi < MTW; mi++) {
        const int mt = mt0 + mi;     // == s
#pragma unroll
        for (int ni = 0; ni < NTB; ni++) {
            const int v = (nt0 + ni) * 16 + n;
#pragma unroll
            for (int r = 0; r < 8; r++) {
                int brow = r + 8 * hi;   // batch index
                out[(size_t)brow * ((size_t)SDIM * VDIM) + (size_t)mt * VDIM + v] = acc[mi][ni][r];
            }
        }
    }
}

// ---------------------------------------------------------------------------
// Host launch
// ---------------------------------------------------------------------------
extern "C" void kernel_launch(void* const* d_in, const int* in_sizes, int n_in,
                              void* d_out, int out_size, void* d_ws, size_t ws_size,
                              hipStream_t stream) {
    const int*   target = (const int*)  d_in[0];
    const float* h0     = (const float*)d_in[1];
    const float* c0     = (const float*)d_in[2];
    const float* emb    = (const float*)d_in[3];
    const float* wih    = (const float*)d_in[4];
    const float* whh    = (const float*)d_in[5];
    const float* bih    = (const float*)d_in[6];
    const float* bhh    = (const float*)d_in[7];
    const float* obias  = (const float*)d_in[8];
    float* out = (float*)d_out;

    char* ws = (char*)d_ws;
    size_t off = 0;
    auto take = [&](size_t bytes) { size_t r = off; off += (bytes + 255) & ~(size_t)255; return r; };

    unsigned short* emb16  = (unsigned short*)(ws + take((size_t)VDIM * HDIM * 2));   // 32.8 MB
    unsigned short* wih16  = (unsigned short*)(ws + take((size_t)G4H  * HDIM * 2));   //  2 MB
    unsigned short* whh16  = (unsigned short*)(ws + take((size_t)G4H  * HDIM * 2));   //  2 MB
    unsigned short* eseq   = (unsigned short*)(ws + take((size_t)BDIM * SDIM * HDIM * 2)); // 4 MB
    unsigned short* hs16   = (unsigned short*)(ws + take((size_t)BDIM * SDIM * HDIM * 2)); // 4 MB
    unsigned short* hApt   = (unsigned short*)(ws + take((size_t)BDIM * HDIM * 2));   // 16 KB
    unsigned short* hBpt   = (unsigned short*)(ws + take((size_t)BDIM * HDIM * 2));   // 16 KB
    float*          bsum   = (float*)         (ws + take((size_t)G4H * 4));           //  8 KB
    float*          xw     = (float*)         (ws + take((size_t)SDIM * NTG * 256 * 4)); // 32 MB
    unsigned int*   ctr    = (unsigned int*)  (ws + take(256));

    k_cvt_bf16<<<4096, 256, 0, stream>>>(emb, emb16, VDIM * HDIM / 2);
    k_cvt_bf16<<<1024, 256, 0, stream>>>(wih, wih16, G4H * HDIM / 2);
    k_cvt_bf16<<<1024, 256, 0, stream>>>(whh, whh16, G4H * HDIM / 2);
    k_bias<<<8, 256, 0, stream>>>(bih, bhh, bsum);
    k_h0<<<32, 256, 0, stream>>>(h0, hApt);
    k_init_ctr<<<1, 64, 0, stream>>>(ctr);
    k_gather<<<BDIM * SDIM, 256, 0, stream>>>(target, emb16, eseq);

    // xW: 256 * 128 tiles, 8 waves/block
    k_xw<<<(SDIM * NTG) / 8, 256, 0, stream>>>(eseq, wih16, bsum, xw);

    // persistent recurrence: 16 blocks x 64 threads = 32 waves (one h-tile each)
    k_lstm<<<16, 64, 0, stream>>>(c0, xw, whh16, hApt, hBpt, hs16, ctr);

    // projection: 16 mt-groups x 500 nt-groups blocks, 64 KB dynamic LDS each
    k_logits<<<(SDIM / MT_PER_BLK) * (VDIM / (NTB * 16)), 256, LOGITS_LDS, stream>>>(
        hs16, emb16, obias, out);
}
